// GraphAttentionLayer_54185307407180
// MI455X (gfx1250) — matrix operand order repaired and checked by reading, not compiled, not run on hardware
//
#include <hip/hip_runtime.h>
#include <hip/hip_bf16.h>

// GAT layer: B=8, N=2048, F_in=F_out=256
#define GB 8
#define GN 2048
#define GF 256
#define NEG_BIG (-9000000000000000.0f)
#define PSTR 136   // padded LDS p-tile row stride (halfs): 68 dwords -> conflict-free b128 reads

typedef __attribute__((ext_vector_type(16))) _Float16 v16h;
typedef __attribute__((ext_vector_type(8)))  _Float16 v8h;
typedef __attribute__((ext_vector_type(4)))  _Float16 v4h;
typedef __attribute__((ext_vector_type(8)))  float    v8f;

// ---------------------------------------------------------------------------
// Kernel 0a: convert h (f32) -> f16, flat
__global__ void gat_cvt_h(const float* __restrict__ h, _Float16* __restrict__ h16, int n) {
    int i = blockIdx.x * blockDim.x + threadIdx.x;
    if (i < n) h16[i] = (_Float16)h[i];
}

// Kernel 0b: W[k][f] (f32) -> Wt16[f][k] (f16)  (transposed for contiguous B-operand loads)
__global__ void gat_cvt_w(const float* __restrict__ W, _Float16* __restrict__ wt16) {
    int k = threadIdx.x;          // 0..255
    int f = blockIdx.x;           // 0..255
    wt16[f * GF + k] = (_Float16)W[k * GF + f];
}

// ---------------------------------------------------------------------------
// Kernel 1: Wh = h @ W via v_wmma_f32_16x16x32_f16.
// grid (N/16, B), block 512 (16 waves). Wave w owns f-tile f0 = 16*w, K=256 in 8 steps.
// Output stored transposed f16: whT[b][f][i]
__global__ __launch_bounds__(512) void gat_gemm_wh(const _Float16* __restrict__ h16,
                                                   const _Float16* __restrict__ wt16,
                                                   _Float16* __restrict__ whT) {
    const int b    = blockIdx.y;
    const int i0   = blockIdx.x * 16;
    const int wave = threadIdx.x >> 5;
    const int lane = threadIdx.x & 31;
    const int hi   = lane >> 4;
    const int ln   = lane & 15;
    const int f0   = wave * 16;

    // A (16x32 f16): lane<16 row=ln holds K=0..15 contiguous; hi lanes K=16..31
    const _Float16* arow = h16 + ((size_t)b * GN + i0 + ln) * GF + hi * 16;
    // B (32x16 f16): lane<16 col n=ln holds K=0..15 contiguous; hi lanes K=16..31
    const _Float16* brow = wt16 + (size_t)(f0 + ln) * GF + hi * 16;

    v8f acc = {};
    #pragma unroll
    for (int ks = 0; ks < GF; ks += 32) {
        v8h a0 = *(const v8h*)(arow + ks);
        v8h a1 = *(const v8h*)(arow + ks + 8);
        v16h a, bm;
        #pragma unroll
        for (int q = 0; q < 8; ++q) { a[q] = a0[q]; a[8 + q] = a1[q]; }
        bm = *(const v16h*)(brow + ks);
        acc = __builtin_amdgcn_wmma_f32_16x16x32_f16(false, a, false, bm,
                                                     (short)0, acc, false, false);
    }
    // C/D layout: VGPR r -> (M = r + 8*hi, col = f0 + ln). Store transposed.
    _Float16* dst = whT + ((size_t)b * GF + f0 + ln) * GN + i0 + hi * 8;
    v8h o;
    #pragma unroll
    for (int r = 0; r < 8; ++r) o[r] = (_Float16)acc[r];
    *(v8h*)dst = o;
}

// ---------------------------------------------------------------------------
// Kernel 2: e1[i] = Wh[i,:]@a1 ; e2[i] = Wh[i,:]@a2  (coalesced column reads of whT)
__global__ __launch_bounds__(256) void gat_e(const _Float16* __restrict__ whT,
                                             const float* __restrict__ avec,
                                             float* __restrict__ e1, float* __restrict__ e2) {
    const int b = blockIdx.y;
    const int i = blockIdx.x * 256 + threadIdx.x;
    const _Float16* base = whT + (size_t)b * GF * GN + i;
    float s1 = 0.f, s2 = 0.f;
    #pragma unroll 4
    for (int f = 0; f < GF; ++f) {
        float v = (float)base[(size_t)f * GN];
        s1 += v * avec[f];
        s2 += v * avec[GF + f];
    }
    e1[b * GN + i] = s1;
    e2[b * GN + i] = s2;
}

// ---------------------------------------------------------------------------
// Kernel 3: per-row softmax stats (max, 1/sumexp) — one wave32 per row,
// int4/float4 streaming + online max/sum + wave32 shfl reduction.
__global__ __launch_bounds__(256) void gat_rowstats(const int* __restrict__ adj,
                                                    const float* __restrict__ e1,
                                                    const float* __restrict__ e2,
                                                    float* __restrict__ rmax,
                                                    float* __restrict__ rsinv) {
    const int b    = blockIdx.y;
    const int row  = blockIdx.x * 8 + (threadIdx.x >> 5);
    const int lane = threadIdx.x & 31;
    const float ev1 = e1[b * GN + row];
    const int* arow = adj + ((size_t)b * GN + row) * GN;
    const float* e2b = e2 + b * GN;

    float m = NEG_BIG, s = 0.f;
    for (int j = lane * 4; j < GN; j += 128) {
        int4   av = *(const int4*)(arow + j);
        float4 ev = *(const float4*)(e2b + j);
        int   ai[4] = {av.x, av.y, av.z, av.w};
        float ef[4] = {ev.x, ev.y, ev.z, ev.w};
        #pragma unroll
        for (int q = 0; q < 4; ++q) {
            float v  = (ai[q] > 0) ? (ev1 + ef[q]) : NEG_BIG;
            float nm = fmaxf(m, v);
            s = s * __expf(m - nm) + __expf(v - nm);
            m = nm;
        }
    }
    #pragma unroll
    for (int off = 16; off > 0; off >>= 1) {
        float m2 = __shfl_xor(m, off, 32);
        float s2 = __shfl_xor(s, off, 32);
        float nm = fmaxf(m, m2);
        s = s * __expf(m - nm) + s2 * __expf(m2 - nm);
        m = nm;
    }
    if (lane == 0) {
        rmax[b * GN + row]  = m;
        rsinv[b * GN + row] = 1.0f / s;
    }
}

// ---------------------------------------------------------------------------
// Kernel 4: out = elu( softmax(mask(e)) @ Wh ), fused.
// grid (N/32, B), block 512 (16 waves). Block owns 32 i-rows x all 256 f-cols.
// Per 128-j chunk: block builds 32x128 f16 probability tile in padded LDS,
// then each wave runs 4 K-steps x 2 row-tiles = 8 WMMAs sharing each B tile.
__global__ __launch_bounds__(512) void gat_attn(const int* __restrict__ adj,
                                                const _Float16* __restrict__ whT,
                                                const float* __restrict__ e1,
                                                const float* __restrict__ e2,
                                                const float* __restrict__ rmax,
                                                const float* __restrict__ rsinv,
                                                float* __restrict__ out) {
    __shared__ _Float16 pt[32 * PSTR];

    const int b    = blockIdx.y;
    const int i0   = blockIdx.x * 32;
    const int tid  = threadIdx.x;
    const int lane = tid & 31;
    const int hi   = lane >> 4;
    const int ln   = lane & 15;
    const int f0   = (tid >> 5) * 16;   // wave's f-tile

    // p-tile worker identity: thread handles rows (prow, prow+16), 4 consecutive cols
    const int prow = tid >> 5;          // 0..15
    const int pc4  = (tid & 31) * 4;    // 0..124
    const float ev1a = e1[b * GN + i0 + prow];
    const float ev1b = e1[b * GN + i0 + prow + 16];
    const float ma   = rmax[b * GN + i0 + prow];
    const float mb   = rmax[b * GN + i0 + prow + 16];
    const float sa   = rsinv[b * GN + i0 + prow];
    const float sb   = rsinv[b * GN + i0 + prow + 16];
    const int* arowa = adj + ((size_t)b * GN + i0 + prow) * GN;
    const int* arowb = arowa + (size_t)16 * GN;
    const float* e2b = e2 + b * GN;

    // B-operand stripe: whT[b][f0+ln][j + hi*16 .. +15]
    const _Float16* brow = whT + ((size_t)b * GF + f0 + ln) * GN + hi * 16;

    v8f acc0 = {}, acc1 = {};
    for (int j0 = 0; j0 < GN; j0 += 128) {
        __syncthreads();  // previous A reads complete before overwrite
        __builtin_prefetch(arowa + j0 + 128 + pc4, 0, 1);
        __builtin_prefetch(arowb + j0 + 128 + pc4, 0, 1);
        int4   ava = *(const int4*)(arowa + j0 + pc4);
        int4   avb = *(const int4*)(arowb + j0 + pc4);
        float4 ev  = *(const float4*)(e2b + j0 + pc4);
        int   aia[4] = {ava.x, ava.y, ava.z, ava.w};
        int   aib[4] = {avb.x, avb.y, avb.z, avb.w};
        float ef[4]  = {ev.x, ev.y, ev.z, ev.w};
        v4h pa, pb;
        #pragma unroll
        for (int q = 0; q < 4; ++q) {
            float va = (aia[q] > 0) ? __expf(ev1a + ef[q] - ma) * sa : 0.0f;
            float vb = (aib[q] > 0) ? __expf(ev1b + ef[q] - mb) * sb : 0.0f;
            pa[q] = (_Float16)va;
            pb[q] = (_Float16)vb;
        }
        *(v4h*)(pt + prow * PSTR + pc4)        = pa;
        *(v4h*)(pt + (prow + 16) * PSTR + pc4) = pb;
        __syncthreads();  // tile visible to all waves

        #pragma unroll
        for (int kk = 0; kk < 4; ++kk) {
            // A layout: lane<16 row=ln holds K in [base..base+15]; hi lanes [base+16..base+31]
            const v8h* p0 = (const v8h*)(pt + ln * PSTR        + kk * 32 + hi * 16);
            const v8h* p1 = (const v8h*)(pt + (16 + ln) * PSTR + kk * 32 + hi * 16);
            v8h a00 = p0[0], a01 = p0[1];
            v8h a10 = p1[0], a11 = p1[1];
            v16h a0, a1, bm;
            #pragma unroll
            for (int q = 0; q < 8; ++q) {
                a0[q] = a00[q]; a0[8 + q] = a01[q];
                a1[q] = a10[q]; a1[8 + q] = a11[q];
            }
            bm = *(const v16h*)(brow + j0 + kk * 32);  // shared by both row-tiles
            acc0 = __builtin_amdgcn_wmma_f32_16x16x32_f16(false, a0, false, bm,
                                                          (short)0, acc0, false, false);
            acc1 = __builtin_amdgcn_wmma_f32_16x16x32_f16(false, a1, false, bm,
                                                          (short)0, acc1, false, false);
        }
    }

    // ELU epilogue: VGPR r -> (row = base + r + 8*hi, col = f0 + ln)
    float* dst0 = out + ((size_t)b * GN + i0 + hi * 8) * GF + f0 + ln;
    float* dst1 = out + ((size_t)b * GN + i0 + 16 + hi * 8) * GF + f0 + ln;
    #pragma unroll
    for (int r = 0; r < 8; ++r) {
        float x0 = acc0[r];
        float x1 = acc1[r];
        dst0[(size_t)r * GF] = (x0 > 0.f) ? x0 : expm1f(x0);
        dst1[(size_t)r * GF] = (x1 > 0.f) ? x1 : expm1f(x1);
    }
}

// ---------------------------------------------------------------------------
extern "C" void kernel_launch(void* const* d_in, const int* in_sizes, int n_in,
                              void* d_out, int out_size, void* d_ws, size_t ws_size,
                              hipStream_t stream) {
    const float* h    = (const float*)d_in[0];   // [B,N,F]
    const int*   adj  = (const int*)  d_in[1];   // [B,N,N]
    const float* W    = (const float*)d_in[2];   // [F,F]
    const float* avec = (const float*)d_in[3];   // [2F,1]
    float* out = (float*)d_out;                  // [B,N,F]

    char* ws = (char*)d_ws;
    _Float16* h16  = (_Float16*)(ws);                              // 8 MB
    _Float16* wt16 = (_Float16*)(ws + 8388608);                    // 128 KB
    _Float16* whT  = (_Float16*)(ws + 8388608 + 131072);           // 8 MB
    float*    e1   = (float*)   (ws + 16908288);                   // 64 KB
    float*    e2   = (float*)   (ws + 16908288 + 65536);           // 64 KB
    float*    rmx  = (float*)   (ws + 16908288 + 131072);          // 64 KB
    float*    rsi  = (float*)   (ws + 16908288 + 196608);          // 64 KB

    const int nH = GB * GN * GF;
    gat_cvt_h<<<(nH + 255) / 256, 256, 0, stream>>>(h, h16, nH);
    gat_cvt_w<<<GF, GF, 0, stream>>>(W, wt16);

    gat_gemm_wh<<<dim3(GN / 16, GB), 512, 0, stream>>>(h16, wt16, whT);

    gat_e<<<dim3(GN / 256, GB), 256, 0, stream>>>(whT, avec, e1, e2);

    gat_rowstats<<<dim3(GN / 8, GB), 256, 0, stream>>>(adj, e1, e2, rmx, rsi);

    gat_attn<<<dim3(GN / 32, GB), 512, 0, stream>>>(adj, whT, e1, e2, rmx, rsi, out);
}
